// PointTransformerCls_40819369181272
// MI455X (gfx1250) — matrix operand order repaired and checked by reading, not compiled, not run on hardware
//
#include <hip/hip_runtime.h>

// ---------------------------------------------------------------------------
// Point Transformer classifier forward for gfx1250 (MI455X), compile-only.
// Dense contractions use v_wmma_f32_16x16x32_f16 (f16 inputs, f32 accum),
// staged through LDS: global_load_b128 -> ds_store_b128 -> ds_load_b128 ->
// v_wmma, with no divergent exec-mask branching anywhere in the hot loop.
// Irregular work (KNN / FPS / gathers / softmax) is VALU + LDS.
// ---------------------------------------------------------------------------

typedef __attribute__((ext_vector_type(16))) _Float16 v16h;
typedef __attribute__((ext_vector_type(8)))  _Float16 v8h;
typedef __attribute__((ext_vector_type(8)))  float    v8f;
typedef __attribute__((ext_vector_type(4)))  float    v4f;

#define BN_EPS 1e-5f
#define NSAMP  16

// ---------------------------------------------------------------------------
// WMMA GEMM:  C[M,N] = A[M,K] @ W[K,N]  (+bias) (+BN) (+ReLU)
// Block = 128 threads (4 waves). Block tile = 32x32 output, K-step = 32.
// Wave (wm, wn) in a 2x2 arrangement computes one 16x16 WMMA tile.
//
// Staging fast path (uniform condition: full tile in-bounds, 16B-aligned
// rows): per thread two global_load_b128 + one ds_store_b128.
// Slow path (ragged K/N = 3, 35, 40, 67, ...): 8 clamped scalar loads whose
// results are zeroed by MULTIPLYING with a 0/1 mask (not a select!) so the
// compiler cannot sink the loads into exec-masked branches; loads batch
// under one wait.
//
// Fragment reads from LDS (ISA 7.12.2 layouts):
//   A: lane -> row = lane&15; halfs K = kgrp..kgrp+7 and 16+kgrp..16+kgrp+7
//      (kgrp = 8*(lane>>4))  => two 16-byte ds reads per fragment.
//   B: lane -> col = lane&15; halfs K = same split as A => 16 strided 2-byte ds reads.
//   C: VGPR j, lane l -> row = j + 8*(l>>4), col = l&15.
// ---------------------------------------------------------------------------
__global__ void __launch_bounds__(128)
gemm_wmma_kernel(const float* __restrict__ A, const float* __restrict__ Wm,
                 const float* __restrict__ bias,
                 const float* __restrict__ bng, const float* __restrict__ bnb,
                 const float* __restrict__ bnm, const float* __restrict__ bnv,
                 float* __restrict__ Cout,
                 int M, int K, int N, int relu)
{
    __shared__ __align__(16) _Float16 Ash[32][32];   // [row][k]
    __shared__ __align__(16) _Float16 Bsh[32][32];   // [k][n]

    const int tid  = threadIdx.x & 127;  // known-bits hint for the compiler
    const int lane = tid & 31;
    const int wave = tid >> 5;
    const int wm   = wave & 1;           // wave row group (0/1)
    const int wn   = wave >> 1;          // wave col group (0/1)
    const int mblk = blockIdx.x * 32;
    const int nblk = blockIdx.y * 32;
    const int kgrp = (lane >> 4) << 3;   // 0 or 8

    const int sr = tid >> 2;             // staging row   (0..31)
    const int sc = (tid & 3) << 3;       // staging col   (0,8,16,24)

    v8f acc = {};

    for (int kt = 0; kt < K; kt += 32) {
        // ---------------- stage A tile ----------------
        const bool fastA = ((K & 3) == 0) && (mblk + 32 <= M) && (kt + 32 <= K);
        if (fastA) {
            const float* ap = &A[(size_t)(mblk + sr) * K + kt + sc];
            v4f f0 = *(const v4f*)ap;
            v4f f1 = *(const v4f*)(ap + 4);
            v8h h;
            #pragma unroll
            for (int e = 0; e < 4; ++e) {
                h[e]     = (_Float16)f0[e];
                h[4 + e] = (_Float16)f1[e];
            }
            *(v8h*)&Ash[sr][sc] = h;
        } else {
            float va[8];
            #pragma unroll
            for (int it = 0; it < 8; ++it) {
                int t = it * 128 + tid;
                int r = t >> 5, k = t & 31;
                int gr = mblk + r, gk = kt + k;
                int cr = gr < M ? gr : M - 1;
                int ck = gk < K ? gk : K - 1;
                float msk = (gr < M && gk < K) ? 1.0f : 0.0f;
                va[it] = A[(size_t)cr * K + ck] * msk;   // load used by fmul -> not sunk
            }
            #pragma unroll
            for (int it = 0; it < 8; ++it) {
                int t = it * 128 + tid;
                Ash[t >> 5][t & 31] = (_Float16)va[it];
            }
        }
        // ---------------- stage B tile ----------------
        const bool fastB = ((N & 3) == 0) && (nblk + 32 <= N) && (kt + 32 <= K);
        if (fastB) {
            const float* bp = &Wm[(size_t)(kt + sr) * N + nblk + sc];
            v4f f0 = *(const v4f*)bp;
            v4f f1 = *(const v4f*)(bp + 4);
            v8h h;
            #pragma unroll
            for (int e = 0; e < 4; ++e) {
                h[e]     = (_Float16)f0[e];
                h[4 + e] = (_Float16)f1[e];
            }
            *(v8h*)&Bsh[sr][sc] = h;
        } else {
            float vb[8];
            #pragma unroll
            for (int it = 0; it < 8; ++it) {
                int t = it * 128 + tid;
                int kr = t >> 5, nc = t & 31;
                int gk = kt + kr, gn = nblk + nc;
                int ck = gk < K ? gk : K - 1;
                int cn = gn < N ? gn : N - 1;
                float msk = (gk < K && gn < N) ? 1.0f : 0.0f;
                vb[it] = Wm[(size_t)ck * N + cn] * msk;  // load used by fmul -> not sunk
            }
            #pragma unroll
            for (int it = 0; it < 8; ++it) {
                int t = it * 128 + tid;
                Bsh[t >> 5][t & 31] = (_Float16)vb[it];
            }
        }
        // uniform-condition prefetch of the next K tile (global_prefetch_b8)
        if (kt + 32 < K) {
            int gr = mblk + sr;
            int cr = gr < M ? gr : M - 1;
            int ck = kt + 32 + sc; ck = ck < K ? ck : K - 1;
            __builtin_prefetch((const void*)&A[(size_t)cr * K + ck], 0, 1);
            int gk = kt + 32 + sr; gk = gk < K ? gk : K - 1;
            int cn = nblk + sc; cn = cn < N ? cn : N - 1;
            __builtin_prefetch((const void*)&Wm[(size_t)gk * N + cn], 0, 1);
        }
        __syncthreads();

        // ---- fragment loads: unconditional LDS vector reads ----
        const int arow = (wm << 4) + (lane & 15);
        const v8h* ap = (const v8h*)&Ash[arow][0];     // 4 x v8h per row
        v8h alo = ap[kgrp >> 3];                       // K = kgrp .. kgrp+7
        v8h ahi = ap[2 + (kgrp >> 3)];                 // K = 16+kgrp .. 16+kgrp+7
        v16h af = __builtin_shufflevector(alo, ahi,
                    0,1,2,3,4,5,6,7,8,9,10,11,12,13,14,15);
        v16h bf;                                       // col = lane&15, K = same order as A
        #pragma unroll
        for (int s = 0; s < 16; ++s)
            bf[s] = Bsh[s < 8 ? kgrp + s : 8 + kgrp + s][(wn << 4) + (lane & 15)];

        acc = __builtin_amdgcn_wmma_f32_16x16x32_f16(
                  false, af, false, bf, (short)0, acc, false, false);
        __syncthreads();
    }

    // ---- epilogue: bias / BN / ReLU fused ----
    const int col   = nblk + (wn << 4) + (lane & 15);
    const int rbase = mblk + (wm << 4) + ((lane >> 4) << 3);
    if (col < N) {
        float bs = bias ? bias[col] : 0.0f;
        float sc2 = 1.0f, sh = 0.0f;
        if (bng) {
            sc2 = bng[col] * rsqrtf(bnv[col] + BN_EPS);
            sh  = bnb[col] - bnm[col] * sc2;
        }
        #pragma unroll
        for (int j = 0; j < 8; ++j) {
            int r = rbase + j;
            if (r < M) {
                float v = acc[j] + bs;
                if (bng) v = v * sc2 + sh;
                if (relu) v = fmaxf(v, 0.0f);
                Cout[(size_t)r * N + col] = v;
            }
        }
    }
}

// ---------------------------------------------------------------------------
// KNN: 16 nearest neighbors excluding self. Thread-per-query, LDS point tiles,
// register-resident insertion sort (fully unrolled -> no scratch).
// ---------------------------------------------------------------------------
__global__ void __launch_bounds__(256)
knn_kernel(const float* __restrict__ p, int* __restrict__ idx, int N)
{
    const int b = blockIdx.y;
    const int q = blockIdx.x * 256 + threadIdx.x;
    const float* pb = p + (size_t)b * N * 3;
    const bool act = (q < N);
    const int cq = act ? q : 0;
    float qx = pb[(size_t)cq*3], qy = pb[(size_t)cq*3+1], qz = pb[(size_t)cq*3+2];

    float bd[NSAMP]; int bi[NSAMP];
    #pragma unroll
    for (int t = 0; t < NSAMP; ++t) { bd[t] = 3.0e38f; bi[t] = 0; }

    __shared__ float tx[256], ty[256], tz[256];
    for (int basej = 0; basej < N; basej += 256) {
        int j = basej + threadIdx.x;
        int cj = j < N ? j : N - 1;
        tx[threadIdx.x] = pb[(size_t)cj*3];
        ty[threadIdx.x] = pb[(size_t)cj*3+1];
        tz[threadIdx.x] = pb[(size_t)cj*3+2];
        __syncthreads();
        int lim = N - basej; if (lim > 256) lim = 256;
        if (act) {
            for (int t = 0; t < lim; ++t) {
                int j2 = basej + t;
                if (j2 == q) continue;
                float dx = tx[t]-qx, dy = ty[t]-qy, dz = tz[t]-qz;
                float d = dx*dx + dy*dy + dz*dz;
                if (d < bd[NSAMP-1]) {
                    float cd = d; int ci = j2;
                    #pragma unroll
                    for (int u = 0; u < NSAMP; ++u) {
                        if (cd < bd[u]) {
                            float td = bd[u]; int ti = bi[u];
                            bd[u] = cd; bi[u] = ci; cd = td; ci = ti;
                        }
                    }
                }
            }
        }
        __syncthreads();
    }
    if (act) {
        #pragma unroll
        for (int t = 0; t < NSAMP; ++t)
            idx[((size_t)b * N + q) * NSAMP + t] = bi[t];
    }
}

// ---------------------------------------------------------------------------
// Farthest point sampling, one block per batch; dist array lives in LDS
// (8192 floats = 32 KB, well within CDNA5's 320 KB per WGP).
// ---------------------------------------------------------------------------
__global__ void __launch_bounds__(1024)
fps_kernel(const float* __restrict__ p, int* __restrict__ sidx, int N, int M)
{
    const int b = blockIdx.x;
    const float* pb = p + (size_t)b * N * 3;
    __shared__ float dist[8192];
    __shared__ float rv[1024];
    __shared__ int   ri[1024];
    __shared__ float cur[3];
    const int tid = threadIdx.x;

    float x0 = pb[0], y0 = pb[1], z0 = pb[2];
    for (int i = tid; i < N; i += 1024) {
        float dx = pb[(size_t)i*3]-x0, dy = pb[(size_t)i*3+1]-y0, dz = pb[(size_t)i*3+2]-z0;
        dist[i] = dx*dx + dy*dy + dz*dz;
    }
    if (tid == 0) sidx[(size_t)b * M] = 0;
    __syncthreads();

    for (int s = 1; s < M; ++s) {
        float best = -1.0f; int besti = 0;
        for (int i = tid; i < N; i += 1024)
            if (dist[i] > best) { best = dist[i]; besti = i; }
        rv[tid] = best; ri[tid] = besti;
        __syncthreads();
        for (int offr = 512; offr > 0; offr >>= 1) {
            if (tid < offr) {
                if (rv[tid+offr] > rv[tid] ||
                    (rv[tid+offr] == rv[tid] && ri[tid+offr] < ri[tid])) {
                    rv[tid] = rv[tid+offr]; ri[tid] = ri[tid+offr];
                }
            }
            __syncthreads();
        }
        int nxt = ri[0];
        if (tid == 0) {
            sidx[(size_t)b * M + s] = nxt;
            cur[0] = pb[(size_t)nxt*3]; cur[1] = pb[(size_t)nxt*3+1]; cur[2] = pb[(size_t)nxt*3+2];
        }
        __syncthreads();
        float cx = cur[0], cy = cur[1], cz = cur[2];
        for (int i = tid; i < N; i += 1024) {
            float dx = pb[(size_t)i*3]-cx, dy = pb[(size_t)i*3+1]-cy, dz = pb[(size_t)i*3+2]-cz;
            float d = dx*dx + dy*dy + dz*dz;
            if (d < dist[i]) dist[i] = d;
        }
        __syncthreads();
    }
}

// ------------------------- elementwise / gather kernels --------------------

__global__ void gather_sampled_kernel(const float* __restrict__ pts,
                                      const int* __restrict__ knn,
                                      const int* __restrict__ sidx,
                                      float* __restrict__ newp,
                                      int* __restrict__ nnind,
                                      int Np, int Mp, int total)
{
    int i = blockIdx.x * 256 + threadIdx.x;
    if (i >= total) return;                   // total = B*M
    int m = i % Mp, b = i / Mp;
    int sj = sidx[(size_t)b * Mp + m];
    #pragma unroll
    for (int c = 0; c < 3; ++c)
        newp[((size_t)b * Mp + m) * 3 + c] = pts[((size_t)b * Np + sj) * 3 + c];
    #pragma unroll
    for (int kk = 0; kk < NSAMP; ++kk)
        nnind[((size_t)b * Mp + m) * NSAMP + kk] = knn[((size_t)b * Np + sj) * NSAMP + kk];
}

__global__ void build_concat_kernel(const float* __restrict__ x,
                                    const float* __restrict__ pts,
                                    const float* __restrict__ newp,
                                    const int* __restrict__ nnind,
                                    float* __restrict__ out,
                                    int Np, int Mp, int Cin, int total)
{
    int i = blockIdx.x * 256 + threadIdx.x;
    if (i >= total) return;                   // total = B*M*16*(3+Cin)
    int Ct = 3 + Cin;
    int c = i % Ct; int t = i / Ct;
    int kk = t % NSAMP; int m = (t / NSAMP) % Mp; int b = t / (NSAMP * Mp);
    int j = nnind[((size_t)b * Mp + m) * NSAMP + kk];
    float v;
    if (c < 3) v = pts[((size_t)b * Np + j) * 3 + c] - newp[((size_t)b * Mp + m) * 3 + c];
    else       v = x[((size_t)b * Np + j) * Cin + (c - 3)];
    out[i] = v;
}

__global__ void maxpool_k_kernel(const float* __restrict__ x, float* __restrict__ out,
                                 int C, int total)
{
    int i = blockIdx.x * 256 + threadIdx.x;
    if (i >= total) return;                   // total = B*M*C
    int c = i % C; int r = i / C;
    float mx = -3.0e38f;
    #pragma unroll
    for (int kk = 0; kk < NSAMP; ++kk)
        mx = fmaxf(mx, x[((size_t)r * NSAMP + kk) * C + c]);
    out[i] = mx;
}

__global__ void rel_pos_kernel(const float* __restrict__ p, const int* __restrict__ ind,
                               float* __restrict__ pr, int Np, int total)
{
    int i = blockIdx.x * 256 + threadIdx.x;
    if (i >= total) return;                   // total = B*N*16*3
    int c = i % 3; int t = i / 3;
    int kk = t % NSAMP; int n = (t / NSAMP) % Np; int b = t / (NSAMP * Np);
    int j = ind[((size_t)b * Np + n) * NSAMP + kk];
    pr[i] = p[((size_t)b * Np + j) * 3 + c] - p[((size_t)b * Np + n) * 3 + c];
}

__global__ void qk_pr_bn_kernel(const float* __restrict__ q, const float* __restrict__ k,
                                const float* __restrict__ pr, const int* __restrict__ ind,
                                const float* __restrict__ g, const float* __restrict__ bb,
                                const float* __restrict__ mm, const float* __restrict__ vv,
                                float* __restrict__ w, int Np, int C, int total)
{
    int i = blockIdx.x * 256 + threadIdx.x;
    if (i >= total) return;                   // total = B*N*16*C
    int c = i % C; int t = i / C;
    int kk = t % NSAMP; int n = (t / NSAMP) % Np; int b = t / (NSAMP * Np);
    int j = ind[((size_t)b * Np + n) * NSAMP + kk];
    float val = q[((size_t)b * Np + n) * C + c] - k[((size_t)b * Np + j) * C + c] + pr[i];
    float sc = g[c] * rsqrtf(vv[c] + BN_EPS);
    val = (val - mm[c]) * sc + bb[c];
    w[i] = fmaxf(val, 0.0f);
}

__global__ void attn_out_kernel(const float* __restrict__ w, const float* __restrict__ vf,
                                const float* __restrict__ pr, const int* __restrict__ ind,
                                float* __restrict__ out, int Np, int C, int total)
{
    int i = blockIdx.x * 256 + threadIdx.x;
    if (i >= total) return;                   // total = B*N*C
    int c = i % C; int n = (i / C) % Np; int b = i / (C * Np);
    size_t base = (((size_t)b * Np + n) * NSAMP) * C + c;
    float e[NSAMP]; float mx = -3.0e38f;
    #pragma unroll
    for (int kk = 0; kk < NSAMP; ++kk) { e[kk] = w[base + (size_t)kk * C]; mx = fmaxf(mx, e[kk]); }
    float s = 0.0f;
    #pragma unroll
    for (int kk = 0; kk < NSAMP; ++kk) { e[kk] = __expf(e[kk] - mx); s += e[kk]; }
    float acc = 0.0f;
    #pragma unroll
    for (int kk = 0; kk < NSAMP; ++kk) {
        int j = ind[((size_t)b * Np + n) * NSAMP + kk];
        acc += (vf[((size_t)b * Np + j) * C + c] + pr[base + (size_t)kk * C]) * e[kk];
    }
    out[i] = acc / s;
}

__global__ void bn_relu_kernel(const float* __restrict__ x,
                               const float* __restrict__ g, const float* __restrict__ b,
                               const float* __restrict__ m, const float* __restrict__ v,
                               float* __restrict__ out, int C, int relu, int total)
{
    int i = blockIdx.x * 256 + threadIdx.x;
    if (i >= total) return;
    int c = i % C;
    float sc = g[c] * rsqrtf(v[c] + BN_EPS);
    float val = (x[i] - m[c]) * sc + b[c];
    if (relu) val = fmaxf(val, 0.0f);
    out[i] = val;
}

__global__ void add_relu_kernel(const float* __restrict__ a, const float* __restrict__ b,
                                float* __restrict__ out, int total)
{
    int i = blockIdx.x * 256 + threadIdx.x;
    if (i >= total) return;
    out[i] = fmaxf(a[i] + b[i], 0.0f);
}

__global__ void meanpool_kernel(const float* __restrict__ x, float* __restrict__ out,
                                int Np, int C, int total)
{
    int i = blockIdx.x * 256 + threadIdx.x;
    if (i >= total) return;                   // total = B*C
    int c = i % C; int b = i / C;
    float s = 0.0f;
    for (int n = 0; n < Np; ++n) s += x[((size_t)b * Np + n) * C + c];
    out[i] = s / (float)Np;
}

// ---------------------------------------------------------------------------
// Host orchestration.
// Parameter leaves assume JAX pytree (sorted-dict-key) flattening:
//   0: p
//   cls: bn.b=1 bn.g=2 bn.m=3 bn.v=4, l1.W=5 l1.b=6, l2.W=7 l2.b=8
//   enc(s): base = 9 + 45*s:
//     W1=+0 W3=+1 bn1=+2..5(b,g,m,v) bn2=+6..9 bn3=+10..13
//     tr (tb=base+14): k.W=+0 k.b=+1 p1.W=+2 p1.b=+3 p2.W=+4 p2.b=+5
//        pbn=+6..9(b,g,m,v) q.W=+10 q.b=+11 v.W=+12 v.b=+13
//        w1.W=+14 w1.b=+15 w2.W=+16 w2.b=+17 wbn1=+18..21 wbn2=+22..25
//     td: W=base+40 bn=base+41..44(b,g,m,v)
// ---------------------------------------------------------------------------
extern "C" void kernel_launch(void* const* d_in, const int* in_sizes, int n_in,
                              void* d_out, int out_size, void* d_ws, size_t ws_size,
                              hipStream_t stream)
{
    (void)in_sizes; (void)n_in; (void)out_size; (void)ws_size;
    const float* P = (const float*)d_in[0];
    auto F = [&](int i) { return (const float*)d_in[i]; };

    const int B = 2, N0 = 8192;
    const int planes[5]  = {32, 64, 128, 256, 512};
    const int strides[5] = {1, 4, 4, 4, 4};

    // ---- workspace carve-out (deterministic bump allocator) ----
    char* wsb = (char*)d_ws; size_t off = 0;
    auto alloc = [&](size_t bytes) -> void* {
        void* r = wsb + off; off += (bytes + 255) & ~(size_t)255; return r;
    };
    const size_t XN  = (size_t)B * N0 * 32;       // max B*N*C feature elems
    float* XA   = (float*)alloc(XN * 4);          // transition-down output / residual
    float* XB   = (float*)alloc(XN * 4);          // pt_block output
    float* X1   = (float*)alloc(XN * 4);
    float* Qb   = (float*)alloc(XN * 4);
    float* Kb   = (float*)alloc(XN * 4);
    float* Vb   = (float*)alloc(XN * 4);
    float* Yb   = (float*)alloc(XN * 4);
    float* Y2b  = (float*)alloc(XN * 4);
    float* Y3b  = (float*)alloc(XN * 4);
    float* PTS0 = (float*)alloc((size_t)B * N0 * 3 * 4);
    float* PTS1 = (float*)alloc((size_t)B * N0 * 3 * 4);
    int*   IDX  = (int*)  alloc((size_t)B * N0 * NSAMP * 4);
    int*   NN2  = (int*)  alloc((size_t)B * (N0/4) * NSAMP * 4);
    int*   SIDX = (int*)  alloc((size_t)B * (N0/4) * 4);
    float* PR3  = (float*)alloc((size_t)B * N0 * NSAMP * 3 * 4);
    float* PR3B = (float*)alloc((size_t)B * N0 * NSAMP * 3 * 4);
    const size_t PRN = (size_t)B * N0 * NSAMP * 32;  // max B*N*K*C elems
    float* PRb  = (float*)alloc(PRN * 4);
    float* WBf  = (float*)alloc(PRN * 4);
    float* WTf  = (float*)alloc(PRN * 4);
    float* CC   = (float*)alloc((size_t)4390912 * 4);  // max B*M*16*(3+Cin)
    float* TDO  = (float*)alloc((size_t)4194304 * 4);  // max B*M*16*C
    float* POOL = (float*)alloc((size_t)B * 512 * 4);
    float* Hc   = (float*)alloc((size_t)B * 512 * 4);

    auto gemm = [&](const float* A, const float* Wt, const float* bias,
                    const float* g, const float* b, const float* m, const float* v,
                    float* Cc, int M, int K, int N, int relu) {
        dim3 grid((M + 31) / 32, (N + 31) / 32);
        gemm_wmma_kernel<<<grid, dim3(128), 0, stream>>>(A, Wt, bias, g, b, m, v,
                                                         Cc, M, K, N, relu);
    };
    auto EW = [&](int total) { return dim3((total + 255) / 256); };

    const float* x   = P; int Cin = 3;
    const float* pts = P; int N = N0;

    for (int s = 0; s < 5; ++s) {
        const int C    = planes[s];
        const int base = 9 + 45 * s;
        const int tb   = base + 14;

        // ---------------- transition down ----------------
        if (strides[s] == 1) {
            gemm(x, F(base+40), nullptr,
                 F(base+42), F(base+41), F(base+43), F(base+44),
                 XA, B * N, Cin, C, 1);
        } else {
            const int M = N / 4;
            fps_kernel<<<dim3(B), dim3(1024), 0, stream>>>(pts, SIDX, N, M);
            float* newp = (pts == PTS0) ? PTS1 : PTS0;
            {
                int total = B * M;
                gather_sampled_kernel<<<EW(total), 256, 0, stream>>>(
                    pts, IDX, SIDX, newp, NN2, N, M, total);
            }
            {
                int Ct = 3 + Cin;
                int total = B * M * NSAMP * Ct;
                build_concat_kernel<<<EW(total), 256, 0, stream>>>(
                    x, pts, newp, NN2, CC, N, M, Cin, total);
                gemm(CC, F(base+40), nullptr,
                     F(base+42), F(base+41), F(base+43), F(base+44),
                     TDO, B * M * NSAMP, Ct, C, 1);
            }
            {
                int total = B * M * C;
                maxpool_k_kernel<<<EW(total), 256, 0, stream>>>(TDO, XA, C, total);
            }
            pts = newp; N = M;
        }

        // ---------------- pt_block ----------------
        gemm(XA, F(base+0), nullptr,
             F(base+3), F(base+2), F(base+4), F(base+5),
             X1, B * N, C, C, 1);                       // relu(bn1(x@W1))

        gemm(X1, F(tb+10), F(tb+11), nullptr, nullptr, nullptr, nullptr,
             Qb, B * N, C, C, 0);                       // q
        gemm(X1, F(tb+0),  F(tb+1),  nullptr, nullptr, nullptr, nullptr,
             Kb, B * N, C, C, 0);                       // k
        gemm(X1, F(tb+12), F(tb+13), nullptr, nullptr, nullptr, nullptr,
             Vb, B * N, C, C, 0);                       // v

        knn_kernel<<<dim3((N + 255) / 256, B), dim3(256), 0, stream>>>(pts, IDX, N);

        { int total = B * N * NSAMP * 3;
          rel_pos_kernel<<<EW(total), 256, 0, stream>>>(pts, IDX, PR3, N, total); }
        gemm(PR3, F(tb+2), F(tb+3),
             F(tb+7), F(tb+6), F(tb+8), F(tb+9),
             PR3B, B * N * NSAMP, 3, 3, 1);             // relu(pbn(p1))
        gemm(PR3B, F(tb+4), F(tb+5), nullptr, nullptr, nullptr, nullptr,
             PRb, B * N * NSAMP, 3, C, 0);              // p2 -> pr

        { int total = B * N * NSAMP * C;
          qk_pr_bn_kernel<<<EW(total), 256, 0, stream>>>(
              Qb, Kb, PRb, IDX,
              F(tb+19), F(tb+18), F(tb+20), F(tb+21),
              WBf, N, C, total); }                      // relu(wbn1(q - k[j] + pr))
        gemm(WBf, F(tb+14), F(tb+15),
             F(tb+23), F(tb+22), F(tb+24), F(tb+25),
             WTf, B * N * NSAMP, C, C, 1);              // relu(wbn2(w1))
        gemm(WTf, F(tb+16), F(tb+17), nullptr, nullptr, nullptr, nullptr,
             WBf, B * N * NSAMP, C, C, 0);              // w2

        { int total = B * N * C;
          attn_out_kernel<<<EW(total), 256, 0, stream>>>(
              WBf, Vb, PRb, IDX, Yb, N, C, total); }    // softmax + aggregate
        { int total = B * N * C;
          bn_relu_kernel<<<EW(total), 256, 0, stream>>>(
              Yb, F(base+7), F(base+6), F(base+8), F(base+9),
              Y2b, C, 1, total); }                      // relu(bn2(y))
        gemm(Y2b, F(base+1), nullptr,
             F(base+11), F(base+10), F(base+12), F(base+13),
             Y3b, B * N, C, C, 0);                      // bn3(y@W3)
        { int total = B * N * C;
          add_relu_kernel<<<EW(total), 256, 0, stream>>>(Y3b, XA, XB, total); }

        x = XB; Cin = C;
    }

    // ---------------- classifier head ----------------
    { int total = B * 512;
      meanpool_kernel<<<EW(total), 256, 0, stream>>>(x, POOL, N, 512, total); }
    gemm(POOL, F(5), F(6),
         F(2), F(1), F(3), F(4),
         Hc, B, 512, 512, 1);                           // relu(bn(l1))
    gemm(Hc, F(7), F(8), nullptr, nullptr, nullptr, nullptr,
         (float*)d_out, B, 512, 40, 0);                 // l2 -> logits
}
